// GraphTransformerLayer_58944131170959
// MI455X (gfx1250) — matrix operand order repaired and checked by reading, not compiled, not run on hardware
//
#include <hip/hip_runtime.h>
#include <math.h>

// ---------------- problem constants ----------------
constexpr int NNODES = 50000;     // 3125 * 16  (exact WMMA row tiles)
constexpr int NEDGES = 800000;    // 8 * 100000 (one wave32 per edge)
constexpr float SCALE = 0.17677669529663687f;  // 32^-0.5

typedef __attribute__((ext_vector_type(16))) __bf16 bf16x16;
typedef __attribute__((ext_vector_type(8)))  float  f32x8;

// ---------------------------------------------------------------------------
// Pack a row-major fp32 weight W[K][srcN] into bf16 WMMA B-fragment order.
// B k order used here (contiguous k, same as the A loads; 32xK x 16xN tile):
//   lane<16 : col N = nt*16 + lane,     K = kt*32 + i      (i = 0..15)
//   lane>=16: col N = nt*16 + lane-16,  K = kt*32 + 16 + i
// Packed element offset = ((nt*Ktiles + kt)*32 + lane)*16 + i  (halves)
// ---------------------------------------------------------------------------
__global__ void pack_w_kernel(const float* __restrict__ W, __bf16* __restrict__ dst,
                              int K, int srcN, int ntBase, int Ktiles)
{
    int idx = blockIdx.x * 256 + threadIdx.x;
    int total = K * srcN;
    if (idx >= total) return;
    int k = idx / srcN;
    int n = idx - k * srcN;
    int kt  = k >> 5;
    int r   = k & 31;
    int hib = r >> 4;
    int i   = r & 15;
    int lane = (n & 15) | (hib << 4);
    int nt   = ntBase + (n >> 4);
    size_t o = (((size_t)nt * Ktiles + kt) * 32 + lane) * 16 + i;
    dst[o] = (__bf16)W[idx];
}

// ---------------------------------------------------------------------------
// WMMA GEMM: C[M x Nout] = act(A[M x K](fp32) * Wpacked(bf16) + bias)
// block = 256 threads = 8 waves; wave w computes a 16x32 output tile
// (two adjacent 16x16 N-tiles) so one A fragment feeds two WMMAs:
//   rows [blockIdx.x*16, +16), cols [(blockIdx.y*8+w)*32, +32).
// act: 0 = none, 1 = exact GELU.
// ---------------------------------------------------------------------------
__global__ __launch_bounds__(256)
void wmma_gemm_kernel(const float* __restrict__ A, int lda,
                      const __bf16* __restrict__ Wp,
                      const float* __restrict__ bias,
                      float* __restrict__ C, int ldc,
                      int Ktiles, int act)
{
    const int lane = threadIdx.x & 31;
    const int wid  = threadIdx.x >> 5;
    const int nt0  = (blockIdx.y * 8 + wid) * 2;   // first of two N-tiles
    const int row0 = blockIdx.x * 16;
    const int hi   = lane >> 4;
    const int rA   = row0 + (lane & 15);

    f32x8 c0 = {};
    f32x8 c1 = {};
    const __bf16* wp0  = Wp + (size_t)nt0 * Ktiles * 512 + (size_t)lane * 16;
    const __bf16* wp1  = wp0 + (size_t)Ktiles * 512;
    const float*  arow = A + (size_t)rA * lda + hi * 16;

    for (int kt = 0; kt < Ktiles; ++kt) {
        const float* ab = arow + kt * 32;
        __builtin_prefetch(ab + 32, 0, 3);   // global_prefetch_b8: next K tile
        float4 f0 = *(const float4*)(ab);
        float4 f1 = *(const float4*)(ab + 4);
        float4 f2 = *(const float4*)(ab + 8);
        float4 f3 = *(const float4*)(ab + 12);
        bf16x16 a;
        a[0]  = (__bf16)f0.x; a[1]  = (__bf16)f0.y; a[2]  = (__bf16)f0.z; a[3]  = (__bf16)f0.w;
        a[4]  = (__bf16)f1.x; a[5]  = (__bf16)f1.y; a[6]  = (__bf16)f1.z; a[7]  = (__bf16)f1.w;
        a[8]  = (__bf16)f2.x; a[9]  = (__bf16)f2.y; a[10] = (__bf16)f2.z; a[11] = (__bf16)f2.w;
        a[12] = (__bf16)f3.x; a[13] = (__bf16)f3.y; a[14] = (__bf16)f3.z; a[15] = (__bf16)f3.w;
        bf16x16 b0 = *(const bf16x16*)(wp0 + (size_t)kt * 512);
        bf16x16 b1 = *(const bf16x16*)(wp1 + (size_t)kt * 512);
        c0 = __builtin_amdgcn_wmma_f32_16x16x32_bf16(false, a, false, b0,
                                                     (short)0, c0, false, false);
        c1 = __builtin_amdgcn_wmma_f32_16x16x32_bf16(false, a, false, b1,
                                                     (short)0, c1, false, false);
    }

    const int col0 = nt0 * 16 + (lane & 15);
    const float bv0 = bias ? bias[col0]      : 0.0f;
    const float bv1 = bias ? bias[col0 + 16] : 0.0f;
#pragma unroll
    for (int r = 0; r < 8; ++r) {
        int row = row0 + r + hi * 8;        // C layout: VGPR r -> M = r (+8 for hi lanes)
        float v0 = c0[r] + bv0;
        float v1 = c1[r] + bv1;
        if (act == 1) {
            v0 = 0.5f * v0 * (1.0f + erff(v0 * 0.70710678118654752f));
            v1 = 0.5f * v1 * (1.0f + erff(v1 * 0.70710678118654752f));
        }
        C[(size_t)row * ldc + col0]      = v0;
        C[(size_t)row * ldc + col0 + 16] = v1;
    }
}

// ---------------------------------------------------------------------------
__global__ void zero_kernel(float* __restrict__ p, size_t n)
{
    size_t i = (size_t)blockIdx.x * 256 + threadIdx.x;
    if (i < n) p[i] = 0.0f;
}

// ---------------------------------------------------------------------------
// Edge pass 1: scores[e][h] = SCALE * dot(Q[dst][h], K[src][h]); scatter-max.
// One wave32 per edge; lane l handles dim l of each of 8 heads; shuffle-reduce.
// attn_max initialized to 0 and reference includes 0 in the max, so the
// running max is always >= 0 -> int-punned atomicMax is exact for positives.
// ---------------------------------------------------------------------------
__global__ __launch_bounds__(256)
void edge_score_kernel(const int* __restrict__ ei, const float* __restrict__ QKV,
                       float* __restrict__ sc, float* __restrict__ amax)
{
    const int lane = threadIdx.x & 31;
    const int e = blockIdx.x * 8 + (threadIdx.x >> 5);
    const int src = ei[e];
    const int dst = ei[NEDGES + e];
    const float* q  = QKV + (size_t)dst * 768;         // Q at cols [0,256)
    const float* kk = QKV + (size_t)src * 768 + 256;   // K at cols [256,512)
    float mine = 0.0f;
#pragma unroll
    for (int h = 0; h < 8; ++h) {
        float p = q[h * 32 + lane] * kk[h * 32 + lane];
#pragma unroll
        for (int m = 16; m; m >>= 1) p += __shfl_xor(p, m, 32);
        if (lane == h) mine = p * SCALE;
    }
    if (lane < 8) {
        sc[(size_t)e * 8 + lane] = mine;
        if (mine > 0.0f)
            atomicMax((int*)&amax[(size_t)dst * 8 + lane], __float_as_int(mine));
    }
}

// Edge pass 2: w = exp(s - max[dst]); scatter-add to attn_sum. 1 thread/(e,h).
__global__ void edge_soft_kernel(const int* __restrict__ ei, float* __restrict__ sc,
                                 const float* __restrict__ amax, float* __restrict__ asum)
{
    size_t t = (size_t)blockIdx.x * 256 + threadIdx.x;
    if (t >= (size_t)NEDGES * 8) return;
    int e = (int)(t >> 3);
    int h = (int)(t & 7);
    int dst = ei[NEDGES + e];
    float w = expf(sc[t] - amax[(size_t)dst * 8 + h]);
    sc[t] = w;
    atomicAdd(&asum[(size_t)dst * 8 + h], w);
}

// Edge pass 3: aggr[dst] += V[src] * w/(sum[dst]+1e-8). One wave32 per edge.
__global__ __launch_bounds__(256)
void edge_aggr_kernel(const int* __restrict__ ei, const float* __restrict__ QKV,
                      const float* __restrict__ sc, const float* __restrict__ asum,
                      float* __restrict__ aggr)
{
    const int lane = threadIdx.x & 31;
    const int e = blockIdx.x * 8 + (threadIdx.x >> 5);
    const int src = ei[e];
    const int dst = ei[NEDGES + e];
    float wn = 0.0f;
    if (lane < 8)
        wn = sc[(size_t)e * 8 + lane] / (asum[(size_t)dst * 8 + lane] + 1e-8f);
    const float* v = QKV + (size_t)src * 768 + 512;    // V at cols [512,768)
    float* o = aggr + (size_t)dst * 256;
#pragma unroll
    for (int h = 0; h < 8; ++h) {
        float wh = __shfl(wn, h, 32);
        atomicAdd(&o[h * 32 + lane], v[h * 32 + lane] * wh);
    }
}

// ---------------------------------------------------------------------------
// out = LayerNorm(xa + xb) * g + b ; one wave32 per 256-wide row.
// ---------------------------------------------------------------------------
__global__ __launch_bounds__(256)
void ln_kernel(const float* __restrict__ xa, const float* __restrict__ xb,
               const float* __restrict__ g, const float* __restrict__ b,
               float* __restrict__ out)
{
    const int lane = threadIdx.x & 31;
    const int row  = blockIdx.x * 8 + (threadIdx.x >> 5);
    const float* pa = xa + (size_t)row * 256;
    const float* pb = xb + (size_t)row * 256;
    float v[8];
    float s = 0.0f;
#pragma unroll
    for (int j = 0; j < 8; ++j) { v[j] = pa[j * 32 + lane] + pb[j * 32 + lane]; s += v[j]; }
#pragma unroll
    for (int m = 16; m; m >>= 1) s += __shfl_xor(s, m, 32);
    float mu = s * (1.0f / 256.0f);
    float ss = 0.0f;
#pragma unroll
    for (int j = 0; j < 8; ++j) { float d = v[j] - mu; ss += d * d; }
#pragma unroll
    for (int m = 16; m; m >>= 1) ss += __shfl_xor(ss, m, 32);
    float rstd = rsqrtf(ss * (1.0f / 256.0f) + 1e-5f);
    float* po = out + (size_t)row * 256;
#pragma unroll
    for (int j = 0; j < 8; ++j) {
        int cidx = j * 32 + lane;
        po[cidx] = (v[j] - mu) * rstd * g[cidx] + b[cidx];
    }
}

// ---------------------------------------------------------------------------
extern "C" void kernel_launch(void* const* d_in, const int* in_sizes, int n_in,
                              void* d_out, int out_size, void* d_ws, size_t ws_size,
                              hipStream_t stream)
{
    const float* x     = (const float*)d_in[0];
    const int*   ei    = (const int*)  d_in[1];
    const float* Wq    = (const float*)d_in[3];
    const float* Wk    = (const float*)d_in[4];
    const float* Wv    = (const float*)d_in[5];
    const float* Wo    = (const float*)d_in[6];
    const float* Wo_b  = (const float*)d_in[7];
    const float* ln1_g = (const float*)d_in[8];
    const float* ln1_b = (const float*)d_in[9];
    const float* ln2_g = (const float*)d_in[10];
    const float* ln2_b = (const float*)d_in[11];
    const float* W1    = (const float*)d_in[12];
    const float* b1    = (const float*)d_in[13];
    const float* W2    = (const float*)d_in[14];
    const float* b2    = (const float*)d_in[15];
    float* out = (float*)d_out;

    char* ws = (char*)d_ws;
    // --- workspace layout (bytes), heavily reused; peak ~257 MB ---
    float*  QKV     = (float*)(ws);                    // N x 768 fused Q|K|V
    float*  proj    = (float*)(ws);                    // reuse after edge phase
    float*  t1      = (float*)(ws +  51200000ull);     // LN1 output
    float*  ffn_out = (float*)(ws + 102400000ull);     // FFN output
    float*  scores  = (float*)(ws + 153600000ull);     // E x 8 scores / weights
    float*  amax    = (float*)(ws + 179200000ull);     // N x 8
    float*  asum    = (float*)(ws + 180800000ull);     // N x 8
    float*  aggr    = (float*)(ws + 182400000ull);     // N x 256 (ends 233.6MB)
    float*  ffn_h   = (float*)(ws + 153600000ull);     // N x 512 (overlays dead edge bufs)
    __bf16* Wqkvp   = (__bf16*)(ws + 256000000ull);    // 768x256 packed
    __bf16* Wop     = (__bf16*)(ws + 256393216ull);    // 256x256 packed
    __bf16* W1p     = (__bf16*)(ws + 256524288ull);    // 256x512 packed
    __bf16* W2p     = (__bf16*)(ws + 256786432ull);    // 512x256 packed

    // 1) pack all weights to bf16 WMMA fragment layout (deterministic, each call)
    pack_w_kernel<<<256, 256, 0, stream>>>(Wq, Wqkvp, 256, 256,  0, 8);
    pack_w_kernel<<<256, 256, 0, stream>>>(Wk, Wqkvp, 256, 256, 16, 8);
    pack_w_kernel<<<256, 256, 0, stream>>>(Wv, Wqkvp, 256, 256, 32, 8);
    pack_w_kernel<<<256, 256, 0, stream>>>(Wo, Wop,   256, 256,  0, 8);
    pack_w_kernel<<<512, 256, 0, stream>>>(W1, W1p,   256, 512,  0, 8);
    pack_w_kernel<<<512, 256, 0, stream>>>(W2, W2p,   512, 256,  0, 16);

    // 2) zero attn_max | attn_sum | aggr (contiguous: 13.6M floats)
    zero_kernel<<<53125, 256, 0, stream>>>(amax, (size_t)NNODES * 272);

    // 3) fused QKV projection: QKV = x @ [Wq|Wk|Wv]  (bf16 WMMA, f32 accum)
    //    block covers 16 rows x 256 cols (8 waves x two 16x16 tiles)
    wmma_gemm_kernel<<<dim3(3125, 3), 256, 0, stream>>>(x, 256, Wqkvp, nullptr,
                                                        QKV, 768, 8, 0);

    // 4) edge attention (QKV is L2-resident: 153.6MB < 192MB)
    edge_score_kernel<<<100000, 256, 0, stream>>>(ei, QKV, scores, amax);
    edge_soft_kernel <<< 25000, 256, 0, stream>>>(ei, scores, amax, asum);
    edge_aggr_kernel <<<100000, 256, 0, stream>>>(ei, QKV, scores, asum, aggr);

    // 5) output projection + bias
    wmma_gemm_kernel<<<dim3(3125, 1), 256, 0, stream>>>(aggr, 256, Wop, Wo_b,
                                                        proj, 256, 8, 0);

    // 6) t1 = LN1(x + proj)
    ln_kernel<<<6250, 256, 0, stream>>>(x, proj, ln1_g, ln1_b, t1);

    // 7) FFN: gelu(t1 @ W1 + b1) @ W2 + b2
    wmma_gemm_kernel<<<dim3(3125, 2), 256, 0, stream>>>(t1, 256, W1p, b1,
                                                        ffn_h, 512, 8, 1);
    wmma_gemm_kernel<<<dim3(3125, 1), 256, 0, stream>>>(ffn_h, 512, W2p, b2,
                                                        ffn_out, 256, 16, 0);

    // 8) out = LN2(t1 + ffn_out)
    ln_kernel<<<6250, 256, 0, stream>>>(t1, ffn_out, ln2_g, ln2_b, out);
}
